// RingAttentionLayer_82892868813316
// MI455X (gfx1250) — matrix operand order repaired and checked by reading, not compiled, not run on hardware
//
#include <hip/hip_runtime.h>

// ---------------------------------------------------------------------------
// RingAttention (single device -> chunk-local causal attention), MI455X gfx1250
// bf16 WMMA everywhere, fp32 softmax/accumulate.
// ---------------------------------------------------------------------------

typedef __bf16 bf16_t;
typedef __attribute__((ext_vector_type(16))) __bf16 v16bf;
typedef __attribute__((ext_vector_type(8)))  __bf16 v8bf;
typedef __attribute__((ext_vector_type(8)))  float  v8f;

#define D_MODEL   1024
#define SEQ       8192
#define NUM_HEADS 16
#define HEAD_DIM  64
#define CHUNK     2048
#define NCHUNK    4

// ---- WMMA helper: D = A(16x32 bf16) x B(32x16 bf16) + C(16x16 f32) ----------
__device__ __forceinline__ v8f wmma_bf16(v16bf a, v16bf b, v8f c) {
    return __builtin_amdgcn_wmma_f32_16x16x32_bf16(
        /*neg_a=*/false, a, /*neg_b=*/false, b,
        /*c_mod=*/(short)0, c, /*reuse_a=*/false, /*reuse_b=*/false);
}

// A fragment (16x32, MxK) from row-major src[row][k], ld in elements.
// ISA layout: lanes 0-15 row=lane, K={0..7,16..23}; lanes 16-31 row=lane-16,
// K={8..15,24..31}.  Two aligned 16B loads per lane.
__device__ __forceinline__ v16bf load_frag_a(const bf16_t* __restrict__ src,
                                             int ld, int row0, int k0) {
    const int lid = threadIdx.x & 31;
    const int r   = lid & 15;
    const int cb  = k0 + ((lid & 16) ? 8 : 0);
    const bf16_t* p = src + (size_t)(row0 + r) * ld + cb;
    v8bf lo = *(const v8bf*)(p);
    v8bf hi = *(const v8bf*)(p + 16);
    return __builtin_shufflevector(lo, hi, 0,1,2,3,4,5,6,7,8,9,10,11,12,13,14,15);
}

// B fragment (32x16, KxN) where B(k,n) = src[n0+n][k0+k] (row-major [N][K]).
// ISA layout: lanes 0-15 N=lane, K={0..7,16..23}; lanes 16-31 N=lane-16, K={8..15,24..31}.
__device__ __forceinline__ v16bf load_frag_b(const bf16_t* __restrict__ src,
                                             int ld, int n0, int k0) {
    const int lid = threadIdx.x & 31;
    const int n   = n0 + (lid & 15);
    const int kb  = k0 + ((lid & 16) ? 8 : 0);
    const bf16_t* p = src + (size_t)n * ld + kb;
    v8bf lo = *(const v8bf*)(p);
    v8bf hi = *(const v8bf*)(p + 16);
    return __builtin_shufflevector(lo, hi, 0,1,2,3,4,5,6,7,8,9,10,11,12,13,14,15);
}

// ---- fp32 -> bf16 convert ---------------------------------------------------
__global__ void cvt_f32_bf16(const float* __restrict__ in,
                             bf16_t* __restrict__ out, int n) {
    int i = blockIdx.x * blockDim.x + threadIdx.x;
    if (i < n) out[i] = (bf16_t)in[i];
}

// ---- fused QKV projection: Y = Xbf @ W^T, head-split outputs ---------------
// grid (M/128, N/64, 3), block 256 (8 waves, M-stacked 16-row tiles).
__global__ __launch_bounds__(256)
void qkv_gemm(const bf16_t* __restrict__ X,
              const bf16_t* __restrict__ Wq, const bf16_t* __restrict__ Wk,
              const bf16_t* __restrict__ Wv,
              bf16_t* __restrict__ Qb, bf16_t* __restrict__ Kb,
              bf16_t* __restrict__ Vt) {
    const int wave = threadIdx.x >> 5;
    const int lid  = threadIdx.x & 31;
    const int m0   = blockIdx.x * 128 + wave * 16;
    const int n0   = blockIdx.y * 64;
    const int which = blockIdx.z;
    const bf16_t* W = (which == 0) ? Wq : ((which == 1) ? Wk : Wv);

    v8f acc[4] = {v8f{}, v8f{}, v8f{}, v8f{}};
    for (int k0 = 0; k0 < D_MODEL; k0 += 32) {
        v16bf a = load_frag_a(X, D_MODEL, m0, k0);
#pragma unroll
        for (int t = 0; t < 4; ++t) {
            v16bf b = load_frag_b(W, D_MODEL, n0 + t * 16, k0);
            acc[t] = wmma_bf16(a, b, acc[t]);
        }
    }
    // scatter into head-split layouts (V transposed for the PV GEMM)
    const int head  = n0 >> 6;               // 64-wide tile == one head
    const int Mbase = (lid & 16) ? 8 : 0;
    const int nlane = lid & 15;
#pragma unroll
    for (int t = 0; t < 4; ++t) {
        const int hd = t * 16 + nlane;
#pragma unroll
        for (int r = 0; r < 8; ++r) {
            const int m     = m0 + Mbase + r;
            const int chunk = m >> 11;
            const int row   = m & (CHUNK - 1);
            const bf16_t val = (bf16_t)acc[t][r];
            const size_t hc = (size_t)(head * NCHUNK + chunk);
            if (which == 2) {
                Vt[(hc * HEAD_DIM + hd) * CHUNK + row] = val;        // [h][c][64][2048]
            } else {
                bf16_t* dst = (which == 0) ? Qb : Kb;                // [h][c][2048][64]
                dst[(hc * CHUNK + row) * HEAD_DIM + hd] = val;
            }
        }
    }
}

// ---- chunk-local causal flash attention ------------------------------------
// grid (2048/128, NCHUNK, NUM_HEADS), block 256; wave owns 16 query rows.
__global__ __launch_bounds__(256)
void attn_kernel(const bf16_t* __restrict__ Qb, const bf16_t* __restrict__ Kb,
                 const bf16_t* __restrict__ Vt, bf16_t* __restrict__ Ob) {
    __shared__ bf16_t pshm[8][16][40];   // per-wave P transpose tile (+pad)

    const int wave  = threadIdx.x >> 5;
    const int lid   = threadIdx.x & 31;
    const int head  = blockIdx.z;
    const int chunk = blockIdx.y;
    const int q0    = blockIdx.x * 128 + wave * 16;

    const size_t hc = (size_t)(head * NCHUNK + chunk);
    const bf16_t* q  = Qb + hc * CHUNK * HEAD_DIM;
    const bf16_t* k  = Kb + hc * CHUNK * HEAD_DIM;
    const bf16_t* vt = Vt + hc * HEAD_DIM * CHUNK;

    const v16bf aq0 = load_frag_a(q, HEAD_DIM, q0, 0);
    const v16bf aq1 = load_frag_a(q, HEAD_DIM, q0, 32);

    v8f o[4] = {v8f{}, v8f{}, v8f{}, v8f{}};
    float mrow[8], lrow[8];
#pragma unroll
    for (int r = 0; r < 8; ++r) { mrow[r] = -__builtin_inff(); lrow[r] = 0.f; }

    const int   Mbase = (lid & 16) ? 8 : 0;
    const int   nlane = lid & 15;
    const float scale = 0.125f;          // 1/sqrt(64)

    for (int j = 0; j <= q0 + 15; j += 32) {      // causal: keys <= query row
        // S(16x32) = q(16x64) @ k(j..j+31)^T, hd contraction split in two
        v8f s0 = wmma_bf16(aq0, load_frag_b(k, HEAD_DIM, j,      0),  v8f{});
        s0     = wmma_bf16(aq1, load_frag_b(k, HEAD_DIM, j,      32), s0);
        v8f s1 = wmma_bf16(aq0, load_frag_b(k, HEAD_DIM, j + 16, 0),  v8f{});
        s1     = wmma_bf16(aq1, load_frag_b(k, HEAD_DIM, j + 16, 32), s1);

#pragma unroll
        for (int r = 0; r < 8; ++r) {
            const int qrow = q0 + Mbase + r;
            const int c0   = j + nlane;
            float v0 = (c0      <= qrow) ? s0[r] * scale : -__builtin_inff();
            float v1 = (c0 + 16 <= qrow) ? s1[r] * scale : -__builtin_inff();
            // row-max across the 16-lane half (row lives across lanes)
            float mx = fmaxf(v0, v1);
            mx = fmaxf(mx, __shfl_xor(mx, 8, 32));
            mx = fmaxf(mx, __shfl_xor(mx, 4, 32));
            mx = fmaxf(mx, __shfl_xor(mx, 2, 32));
            mx = fmaxf(mx, __shfl_xor(mx, 1, 32));
            const float mnew = fmaxf(mrow[r], mx);
            const float corr = __expf(mrow[r] - mnew);   // 0 on first block
            const float p0 = __expf(v0 - mnew);
            const float p1 = __expf(v1 - mnew);
            float sum = p0 + p1;
            sum += __shfl_xor(sum, 8, 32);
            sum += __shfl_xor(sum, 4, 32);
            sum += __shfl_xor(sum, 2, 32);
            sum += __shfl_xor(sum, 1, 32);
            lrow[r] = lrow[r] * corr + sum;
            mrow[r] = mnew;
#pragma unroll
            for (int t = 0; t < 4; ++t) o[t][r] *= corr;
            // stash P (C layout) into LDS for the A-layout reload
            pshm[wave][Mbase + r][nlane]      = (bf16_t)p0;
            pshm[wave][Mbase + r][16 + nlane] = (bf16_t)p1;
        }
        // reload P as a 16x32 A fragment (wave-private tile; DS is in-order
        // within a wave so no barrier is required)
        v16bf aP;
        {
            const int prow = lid & 15;
            const int cb   = (lid & 16) ? 8 : 0;
#pragma unroll
            for (int i = 0; i < 8; ++i) aP[i]     = pshm[wave][prow][cb + i];
#pragma unroll
            for (int i = 0; i < 8; ++i) aP[8 + i] = pshm[wave][prow][cb + 16 + i];
        }
        // O(16x64) += P(16x32) @ V(32x64); Vt rows are hd -> contiguous keys
#pragma unroll
        for (int t = 0; t < 4; ++t) {
            v16bf bv = load_frag_b(vt, CHUNK, t * 16, j);
            o[t] = wmma_bf16(aP, bv, o[t]);
        }
    }
    // normalize and emit attention output as [seq][d_model] bf16
#pragma unroll
    for (int t = 0; t < 4; ++t) {
#pragma unroll
        for (int r = 0; r < 8; ++r) {
            const float val = o[t][r] / lrow[r];
            const int m   = chunk * CHUNK + q0 + Mbase + r;
            const int col = head * HEAD_DIM + t * 16 + nlane;
            Ob[(size_t)m * D_MODEL + col] = (bf16_t)val;
        }
    }
}

// ---- output projection: out = Ob @ Wo^T (fp32 result) ----------------------
__global__ __launch_bounds__(256)
void out_gemm(const bf16_t* __restrict__ A, const bf16_t* __restrict__ W,
              float* __restrict__ Y) {
    const int wave = threadIdx.x >> 5;
    const int lid  = threadIdx.x & 31;
    const int m0   = blockIdx.x * 128 + wave * 16;
    const int n0   = blockIdx.y * 64;

    v8f acc[4] = {v8f{}, v8f{}, v8f{}, v8f{}};
    for (int k0 = 0; k0 < D_MODEL; k0 += 32) {
        v16bf a = load_frag_a(A, D_MODEL, m0, k0);
#pragma unroll
        for (int t = 0; t < 4; ++t) {
            v16bf b = load_frag_b(W, D_MODEL, n0 + t * 16, k0);
            acc[t] = wmma_bf16(a, b, acc[t]);
        }
    }
    const int Mbase = (lid & 16) ? 8 : 0;
    const int nlane = lid & 15;
#pragma unroll
    for (int t = 0; t < 4; ++t)
#pragma unroll
        for (int r = 0; r < 8; ++r)
            Y[(size_t)(m0 + Mbase + r) * D_MODEL + n0 + t * 16 + nlane] = acc[t][r];
}

// ---------------------------------------------------------------------------
extern "C" void kernel_launch(void* const* d_in, const int* in_sizes, int n_in,
                              void* d_out, int out_size, void* d_ws, size_t ws_size,
                              hipStream_t stream) {
    (void)in_sizes; (void)n_in; (void)out_size; (void)ws_size;
    const float* hidden = (const float*)d_in[0];
    const float* Wq = (const float*)d_in[1];
    const float* Wk = (const float*)d_in[2];
    const float* Wv = (const float*)d_in[3];
    const float* Wo = (const float*)d_in[4];

    const size_t nAct = (size_t)SEQ * D_MODEL;       // 8M elems
    const size_t nW   = (size_t)D_MODEL * D_MODEL;   // 1M elems

    char* ws = (char*)d_ws;
    size_t off = 0;
    bf16_t* xbf = (bf16_t*)(ws + off); off += nAct * sizeof(bf16_t);
    bf16_t* wqb = (bf16_t*)(ws + off); off += nW * sizeof(bf16_t);
    bf16_t* wkb = (bf16_t*)(ws + off); off += nW * sizeof(bf16_t);
    bf16_t* wvb = (bf16_t*)(ws + off); off += nW * sizeof(bf16_t);
    bf16_t* wob = (bf16_t*)(ws + off); off += nW * sizeof(bf16_t);
    bf16_t* qb  = (bf16_t*)(ws + off); off += nAct * sizeof(bf16_t);
    bf16_t* kb  = (bf16_t*)(ws + off); off += nAct * sizeof(bf16_t);
    bf16_t* vt  = (bf16_t*)(ws + off); off += nAct * sizeof(bf16_t);
    bf16_t* ob  = (bf16_t*)(ws + off); off += nAct * sizeof(bf16_t);

    cvt_f32_bf16<<<(int)((nAct + 255) / 256), 256, 0, stream>>>(hidden, xbf, (int)nAct);
    cvt_f32_bf16<<<(int)((nW + 255) / 256), 256, 0, stream>>>(Wq, wqb, (int)nW);
    cvt_f32_bf16<<<(int)((nW + 255) / 256), 256, 0, stream>>>(Wk, wkb, (int)nW);
    cvt_f32_bf16<<<(int)((nW + 255) / 256), 256, 0, stream>>>(Wv, wvb, (int)nW);
    cvt_f32_bf16<<<(int)((nW + 255) / 256), 256, 0, stream>>>(Wo, wob, (int)nW);

    qkv_gemm<<<dim3(SEQ / 128, D_MODEL / 64, 3), 256, 0, stream>>>(
        xbf, wqb, wkb, wvb, qb, kb, vt);

    attn_kernel<<<dim3(CHUNK / 128, NCHUNK, NUM_HEADS), 256, 0, stream>>>(
        qb, kb, vt, ob);

    out_gemm<<<dim3(SEQ / 128, D_MODEL / 64), 256, 0, stream>>>(
        ob, wob, (float*)d_out);
}
